// Net_63531156242919
// MI455X (gfx1250) — compile-verified, matrix operand order checked by reading, not run on hardware
//
#include <hip/hip_runtime.h>

// SNN (4-layer LIF MLP), B=256, T=25, D_IN=768, H=2048, OUT=10.
// bf16 WMMA GEMMs with f32 accumulation + fused LIF epilogue.

typedef __attribute__((ext_vector_type(16))) __bf16 v16bf;
typedef __attribute__((ext_vector_type(8)))  __bf16 bf16x8;
typedef __attribute__((ext_vector_type(8)))  float  v8f;

#define B_ 256
#define T_ 25
#define DIN_ 768
#define H_ 2048
#define OUT_ 10

// ---------------------------------------------------------------------------
// f32 -> bf16 conversion kernels
// ---------------------------------------------------------------------------
__global__ void f32_to_bf16_kernel(const float* __restrict in, __bf16* __restrict out, int n) {
    int i = blockIdx.x * blockDim.x + threadIdx.x;
    if (i < n) out[i] = (__bf16)in[i];
}

// x is [B, T, DIN]; produce time-major bf16 [T, B, DIN] so each timestep's
// activation matrix is contiguous row-major [256 x 768].
__global__ void convert_x_kernel(const float* __restrict x, __bf16* __restrict xbf, int n) {
    int i = blockIdx.x * blockDim.x + threadIdx.x;
    if (i >= n) return;
    int d   = i % DIN_;
    int rem = i / DIN_;
    int b   = rem % B_;
    int t   = rem / B_;
    xbf[i] = (__bf16)x[((size_t)b * T_ + t) * DIN_ + d];
}

// ---------------------------------------------------------------------------
// Fragment loader: 16x32 bf16 tile, rows r0..r0+15, K-slice k0..k0+31.
// Matches the CDNA5 16-bit A-matrix layout (ISA 7.12.2):
//   lanes 0-15  : row = r0+lane,    K = k0+0..7  (v0-3), k0+16..23 (v4-7)
//   lanes 16-31 : row = r0+lane-16, K = k0+8..15 (v0-3), k0+24..31 (v4-7)
// The same loader serves the B operand of A·W^T: B[k][n] = W[n][k], and the
// B-fragment layout over (n,k) equals the A layout over (m,k) -> two
// contiguous 16B loads from row n of row-major W.
// ---------------------------------------------------------------------------
__device__ inline v16bf load_tile16x32(const __bf16* __restrict base, int ld,
                                       int r0, int k0, int lane) {
    const int half = lane >> 4;
    const int r    = lane & 15;
    const __bf16* p = base + (size_t)(r0 + r) * ld + (k0 + half * 8);
    bf16x8 lo = *(const bf16x8*)(p);
    bf16x8 hi = *(const bf16x8*)(p + 16);
    v16bf f;
#pragma unroll
    for (int i = 0; i < 8; ++i) { f[i] = lo[i]; f[i + 8] = hi[i]; }
    return f;
}

// ---------------------------------------------------------------------------
// Fused GEMM (C = A @ W^T + bias) + LIF update.
// A: bf16 [256 x K] row-major, W: bf16 [N x K] row-major, N multiple of 64.
// Block = 4 waves (128 threads), tile 64(M) x 64(N); each wave: 16(M) x 64(N)
// = 4 accumulators of v_wmma_f32_16x16x32_bf16.
// Epilogue per element:
//   c = acc + bias;  mp = (t==0 ? 0 : m_state);
//   reset = (mp - thr > 0); m = clamp(beta,0,1)*mp + c - reset*thr;
//   s = (m - thr > 0)
// c/m/s streamed to d_out with non-temporal stores; m_state and bf16 spikes
// (consumed by the next layer / next timestep) use cached stores.
// ---------------------------------------------------------------------------
__global__ __launch_bounds__(128)
void gemm_lif_kernel(const __bf16* __restrict A,
                     const __bf16* __restrict W,
                     const float*  __restrict bias,
                     float*        __restrict m_state,   // [256 x N]
                     float*        __restrict c_out,     // already + t*B*N
                     float*        __restrict m_out,
                     float*        __restrict s_out,
                     __bf16*       __restrict s_bf,      // [256 x N]
                     const float*  __restrict beta_p,
                     const float*  __restrict thr_p,
                     int K, int N, int t)
{
    const int lane = threadIdx.x & 31;
    const int wave = threadIdx.x >> 5;
    const int col0 = blockIdx.x * 64;
    const int row0 = blockIdx.y * 64 + wave * 16;

    v8f acc[4];
#pragma unroll
    for (int j = 0; j < 4; ++j)
#pragma unroll
        for (int i = 0; i < 8; ++i) acc[j][i] = 0.0f;

    for (int k0 = 0; k0 < K; k0 += 32) {
        v16bf a = load_tile16x32(A, K, row0, k0, lane);
#pragma unroll
        for (int j = 0; j < 4; ++j) {
            v16bf b = load_tile16x32(W, K, col0 + j * 16, k0, lane);
            acc[j] = __builtin_amdgcn_wmma_f32_16x16x32_bf16(
                false, a, false, b, (short)0, acc[j], false, false);
        }
    }

    const float beta = fminf(fmaxf(beta_p[0], 0.0f), 1.0f);
    const float thr  = thr_p[0];

    // C/D layout: lanes 0-15 -> N=lane, M=v; lanes 16-31 -> N=lane-16, M=v+8.
    const int nloc  = lane & 15;
    const int mbase = (lane >> 4) * 8;

#pragma unroll
    for (int j = 0; j < 4; ++j) {
        const int   gc = col0 + j * 16 + nloc;
        const float bv = bias[gc];
#pragma unroll
        for (int v = 0; v < 8; ++v) {
            const int    gr  = row0 + mbase + v;
            const size_t idx = (size_t)gr * N + gc;
            float c  = acc[j][v] + bv;
            float mp = (t == 0) ? 0.0f : m_state[idx];
            float reset = (mp - thr) > 0.0f ? 1.0f : 0.0f;
            float m  = beta * mp + c - reset * thr;
            float s  = (m - thr) > 0.0f ? 1.0f : 0.0f;
            __builtin_nontemporal_store(c, c_out + idx);
            __builtin_nontemporal_store(m, m_out + idx);
            __builtin_nontemporal_store(s, s_out + idx);
            m_state[idx] = m;
            s_bf[idx]    = (__bf16)s;
        }
    }
}

// ---------------------------------------------------------------------------
// Layer 4: N=10 is too skinny for WMMA. One wave per output element
// (256*10 = 2560 waves): lane-strided bf16x8 dot product over K=2048,
// shuffle reduction, LIF on lane 0.
// ---------------------------------------------------------------------------
__global__ __launch_bounds__(256)
void fc_small_lif_kernel(const __bf16* __restrict A,   // [256 x K]
                         const __bf16* __restrict W,   // [N x K]
                         const float*  __restrict bias,
                         float*        __restrict m_state, // [256 x N]
                         float*        __restrict c_out,
                         float*        __restrict m_out,
                         float*        __restrict s_out,
                         const float*  __restrict beta_p,
                         const float*  __restrict thr_p,
                         int K, int N, int t)
{
    const int lane = threadIdx.x & 31;
    const int gw   = (int)((blockIdx.x * blockDim.x + threadIdx.x) >> 5);
    const int row  = gw / N;
    const int o    = gw % N;
    if (row >= B_) return;

    const __bf16* a = A + (size_t)row * K;
    const __bf16* w = W + (size_t)o * K;
    float sum = 0.0f;
    for (int k = lane * 8; k < K; k += 32 * 8) {
        bf16x8 av = *(const bf16x8*)(a + k);
        bf16x8 wv = *(const bf16x8*)(w + k);
#pragma unroll
        for (int i = 0; i < 8; ++i) sum += (float)av[i] * (float)wv[i];
    }
#pragma unroll
    for (int off = 16; off > 0; off >>= 1) sum += __shfl_down(sum, off, 32);

    if (lane == 0) {
        const float beta = fminf(fmaxf(beta_p[0], 0.0f), 1.0f);
        const float thr  = thr_p[0];
        const size_t idx = (size_t)row * N + o;
        float c  = sum + bias[o];
        float mp = (t == 0) ? 0.0f : m_state[idx];
        float reset = (mp - thr) > 0.0f ? 1.0f : 0.0f;
        float m  = beta * mp + c - reset * thr;
        float s  = (m - thr) > 0.0f ? 1.0f : 0.0f;
        __builtin_nontemporal_store(c, c_out + idx);
        __builtin_nontemporal_store(m, m_out + idx);
        __builtin_nontemporal_store(s, s_out + idx);
        m_state[idx] = m;
    }
}

// ---------------------------------------------------------------------------
// Host driver
// ---------------------------------------------------------------------------
extern "C" void kernel_launch(void* const* d_in, const int* in_sizes, int n_in,
                              void* d_out, int out_size, void* d_ws, size_t ws_size,
                              hipStream_t stream) {
    (void)in_sizes; (void)n_in; (void)out_size; (void)ws_size;

    const float* x    = (const float*)d_in[0];
    const float* W1   = (const float*)d_in[1];
    const float* b1   = (const float*)d_in[2];
    const float* W2   = (const float*)d_in[3];
    const float* b2   = (const float*)d_in[4];
    const float* W3   = (const float*)d_in[5];
    const float* b3   = (const float*)d_in[6];
    const float* W4   = (const float*)d_in[7];
    const float* b4   = (const float*)d_in[8];
    const float* beta1 = (const float*)d_in[9];
    const float* beta2 = (const float*)d_in[10];
    const float* beta3 = (const float*)d_in[11];
    const float* beta4 = (const float*)d_in[12];
    const float* thr1  = (const float*)d_in[13];
    const float* thr2  = (const float*)d_in[14];
    const float* thr3  = (const float*)d_in[15];
    const float* thr4  = (const float*)d_in[16];

    float* out = (float*)d_out;
    const size_t Sb = (size_t)T_ * B_ * H_;    // 13,107,200
    const size_t Ss = (size_t)T_ * B_ * OUT_;  // 64,000
    float* s1o = out + 0 * Sb;
    float* s2o = out + 1 * Sb;
    float* s3o = out + 2 * Sb;
    float* s4o = out + 3 * Sb;
    float* m1o = out + 3 * Sb + 1 * Ss;
    float* m2o = out + 4 * Sb + 1 * Ss;
    float* m3o = out + 5 * Sb + 1 * Ss;
    float* m4o = out + 6 * Sb + 1 * Ss;
    float* c1o = out + 6 * Sb + 2 * Ss;
    float* c2o = out + 7 * Sb + 2 * Ss;
    float* c3o = out + 8 * Sb + 2 * Ss;
    float* c4o = out + 9 * Sb + 2 * Ss;

    // Workspace carve-up (256B aligned): ~39.2 MB total.
    char* ws = (char*)d_ws;
    auto carve = [&](size_t bytes) -> char* {
        char* p = ws;
        ws += (bytes + 255) & ~(size_t)255;
        return p;
    };
    __bf16* xbf  = (__bf16*)carve((size_t)T_ * B_ * DIN_ * 2);
    __bf16* w1bf = (__bf16*)carve((size_t)H_ * DIN_ * 2);
    __bf16* w2bf = (__bf16*)carve((size_t)H_ * H_ * 2);
    __bf16* w3bf = (__bf16*)carve((size_t)H_ * H_ * 2);
    __bf16* w4bf = (__bf16*)carve((size_t)OUT_ * H_ * 2);
    __bf16* s1bf = (__bf16*)carve((size_t)B_ * H_ * 2);
    __bf16* s2bf = (__bf16*)carve((size_t)B_ * H_ * 2);
    __bf16* s3bf = (__bf16*)carve((size_t)B_ * H_ * 2);
    float*  m1s  = (float*)carve((size_t)B_ * H_ * 4);
    float*  m2s  = (float*)carve((size_t)B_ * H_ * 4);
    float*  m3s  = (float*)carve((size_t)B_ * H_ * 4);
    float*  m4s  = (float*)carve((size_t)B_ * OUT_ * 4);

    // One-shot conversions (re-done every launch: deterministic, no caching).
    {
        int n = T_ * B_ * DIN_;
        convert_x_kernel<<<(n + 255) / 256, 256, 0, stream>>>(x, xbf, n);
    }
    {
        int n = H_ * DIN_;
        f32_to_bf16_kernel<<<(n + 255) / 256, 256, 0, stream>>>(W1, w1bf, n);
    }
    {
        int n = H_ * H_;
        f32_to_bf16_kernel<<<(n + 255) / 256, 256, 0, stream>>>(W2, w2bf, n);
        f32_to_bf16_kernel<<<(n + 255) / 256, 256, 0, stream>>>(W3, w3bf, n);
    }
    {
        int n = OUT_ * H_;
        f32_to_bf16_kernel<<<(n + 255) / 256, 256, 0, stream>>>(W4, w4bf, n);
    }

    dim3 ggrid(H_ / 64, B_ / 64);   // (32, 4) = 128 workgroups per GEMM
    const int fc_waves  = B_ * OUT_;                 // 2560
    const int fc_blocks = (fc_waves * 32 + 255) / 256;

    for (int t = 0; t < T_; ++t) {
        const size_t ofs  = (size_t)t * B_ * H_;
        const size_t ofs4 = (size_t)t * B_ * OUT_;

        gemm_lif_kernel<<<ggrid, 128, 0, stream>>>(
            xbf + (size_t)t * B_ * DIN_, w1bf, b1, m1s,
            c1o + ofs, m1o + ofs, s1o + ofs, s1bf, beta1, thr1, DIN_, H_, t);

        gemm_lif_kernel<<<ggrid, 128, 0, stream>>>(
            s1bf, w2bf, b2, m2s,
            c2o + ofs, m2o + ofs, s2o + ofs, s2bf, beta2, thr2, H_, H_, t);

        gemm_lif_kernel<<<ggrid, 128, 0, stream>>>(
            s2bf, w3bf, b3, m3s,
            c3o + ofs, m3o + ofs, s3o + ofs, s3bf, beta3, thr3, H_, H_, t);

        fc_small_lif_kernel<<<fc_blocks, 256, 0, stream>>>(
            s3bf, w4bf, b4, m4s,
            c4o + ofs4, m4o + ofs4, s4o + ofs4, beta4, thr4, H_, OUT_, t);
    }
}
